// DirectVoxGO_23373212025332
// MI455X (gfx1250) — matrix operand order repaired and checked by reading, not compiled, not run on hardware
//
#include <hip/hip_runtime.h>
#include <math.h>

typedef __attribute__((ext_vector_type(16))) _Float16 v16h;
typedef __attribute__((ext_vector_type(8)))  float    v8f;

#define RES   160
#define RES2  (RES*RES)
#define RES3  (RES*RES*RES)
#define NRAYS 4096
#define MTOT  1048576
#define ACT_SHIFT (-13.815510557964274f)

// ---------------- WMMA fragment helpers (CDNA5 16x16x32 f16 layouts) ----------------

// A-matrix 16x32 f16: lane<16 -> M=lane, halves 0..15 = K(kb..kb+15), kb = 16*(lane>=16)
__device__ inline v16h load_a_frag(const _Float16* base, int stride, int row0, int kt, int lane) {
  int m  = row0 + (lane & 15);
  int kb = kt * 32 + ((lane >> 4) & 1) * 16;
  const _Float16* p = base + (size_t)m * stride + kb;
  union { v16h v; uint4 q[2]; } u;
  u.q[0] = *(const uint4*)(p);
  u.q[1] = *(const uint4*)(p + 8);
  return u.v;
}

// B-matrix 32x16 f16, stored pre-swizzled: element (k,n) at ((kt*2+g)*N + n)*16 + h, g=k%32/16, h=k%16
// -> one lane's fragment is 16 contiguous halves (32B)
__device__ inline v16h load_b_frag(const _Float16* w, int N, int kt, int nt, int lane) {
  int g = (lane >> 4) & 1, n = lane & 15;
  const _Float16* p = w + ((size_t)((kt * 2 + g) * N + nt * 16 + n) << 4);
  union { v16h v; uint4 q[2]; } u;
  u.q[0] = *(const uint4*)(p);
  u.q[1] = *(const uint4*)(p + 8);
  return u.v;
}

__device__ inline int frag_idx(int k, int n, int N) {
  int kt = k >> 5, kr = k & 31, g = kr >> 4, h = kr & 15;
  return (((kt * 2 + g) * N + n) << 4) + h;
}

// ---------------- trilinear ----------------
__device__ inline float tri8(const float* __restrict__ g, int i000, float fx, float fy, float fz) {
  const int dx = RES2, dy = RES;
  float c000 = g[i000],           c001 = g[i000 + 1];
  float c010 = g[i000 + dy],      c011 = g[i000 + dy + 1];
  float c100 = g[i000 + dx],      c101 = g[i000 + dx + 1];
  float c110 = g[i000 + dx + dy], c111 = g[i000 + dx + dy + 1];
  float c00 = c000 * (1.f - fz) + c001 * fz;
  float c01 = c010 * (1.f - fz) + c011 * fz;
  float c10 = c100 * (1.f - fz) + c101 * fz;
  float c11 = c110 * (1.f - fz) + c111 * fz;
  float c0 = c00 * (1.f - fy) + c01 * fy;
  float c1 = c10 * (1.f - fy) + c11 * fy;
  return c0 * (1.f - fx) + c1 * fx;
}

// ---------------- kernels ----------------

__global__ void init_out_kernel(float* __restrict__ out) {
  int i = blockIdx.x * 256 + threadIdx.x;
  if (i < NRAYS * 3) out[i] = 1.0f;   // alphainv_last default for empty rays
}

__global__ void vemb_kernel(const float* __restrict__ viewdirs, float* __restrict__ vemb) {
  int r = blockIdx.x * 256 + threadIdx.x;
  if (r >= NRAYS) return;
  float v[3] = { viewdirs[r * 3 + 0], viewdirs[r * 3 + 1], viewdirs[r * 3 + 2] };
  float* o = vemb + (size_t)r * 27;
  o[0] = v[0]; o[1] = v[1]; o[2] = v[2];
#pragma unroll
  for (int c = 0; c < 3; ++c)
#pragma unroll
    for (int f = 0; f < 4; ++f) {
      float ph = v[c] * (float)(1 << f);
      o[3  + c * 4 + f] = __sinf(ph);
      o[15 + c * 4 + f] = __cosf(ph);
    }
}

__global__ void prep_weights_kernel(const float* __restrict__ w0, const float* __restrict__ w1,
                                    const float* __restrict__ w2,
                                    _Float16* __restrict__ w0p, _Float16* __restrict__ w1p,
                                    _Float16* __restrict__ w2p) {
  int i = blockIdx.x * 256 + threadIdx.x;
  if (i < 64 * 128) {                                   // w0: K=64(pad from 39), N=128
    int k = i >> 7, n = i & 127;
    float v = (k < 39) ? w0[k * 128 + n] : 0.f;
    w0p[frag_idx(k, n, 128)] = (_Float16)v;
  } else if (i < 64 * 128 + 128 * 128) {                // w1: K=128, N=128
    int j = i - 8192; int k = j >> 7, n = j & 127;
    w1p[frag_idx(k, n, 128)] = (_Float16)w1[k * 128 + n];
  } else if (i < 64 * 128 + 128 * 128 + 128 * 16) {     // w2: K=128, N=16(pad from 3)
    int j = i - 24576; int k = j >> 4, n = j & 15;
    float v = (n < 3) ? w2[k * 3 + n] : 0.f;
    w2p[frag_idx(k, n, 16)] = (_Float16)v;
  }
}

__global__ void sample_kernel(const float* __restrict__ xyz, const float* __restrict__ dgrid,
                              const float* __restrict__ k0grid, const int* __restrict__ ray_id,
                              const float* __restrict__ vemb,
                              _Float16* __restrict__ feat, float* __restrict__ alpha,
                              float* __restrict__ logt) {
  int i = blockIdx.x * 256 + threadIdx.x;
  if (i >= MTOT) return;
  float px = xyz[i * 3 + 0] * (RES - 1);
  float py = xyz[i * 3 + 1] * (RES - 1);
  float pz = xyz[i * 3 + 2] * (RES - 1);
  int ix = (int)floorf(px), iy = (int)floorf(py), iz = (int)floorf(pz);
  ix = min(max(ix, 0), RES - 2); iy = min(max(iy, 0), RES - 2); iz = min(max(iz, 0), RES - 2);
  float fx = px - (float)ix, fy = py - (float)iy, fz = pz - (float)iz;
  int i000 = (ix * RES + iy) * RES + iz;

  float d = tri8(dgrid, i000, fx, fy, fz);
  float a = 1.0f - powf(1.0f + expf(d + ACT_SHIFT), -0.5f);
  alpha[i] = a;
  float acl = fminf(fmaxf(a, 0.0f), 1.0f - 1e-7f);
  logt[i] = log1pf(-acl);

  _Float16* fr = feat + (size_t)i * 64;
#pragma unroll
  for (int c = 0; c < 12; ++c)
    fr[c] = (_Float16)tri8(k0grid + (size_t)c * RES3, i000, fx, fy, fz);
  const float* ve = vemb + (size_t)ray_id[i] * 27;
#pragma unroll
  for (int j = 0; j < 27; ++j) fr[12 + j] = (_Float16)ve[j];
#pragma unroll
  for (int j = 39; j < 64; ++j) fr[j] = (_Float16)0.0f;
}

// ---- 3-phase segmented scan over log_t (ray_id sorted) ----
__global__ void scan1_kernel(const float* __restrict__ logt, float* __restrict__ c,
                             float* __restrict__ bsums) {
  __shared__ float s[256];
  int t = threadIdx.x;
  size_t base = (size_t)blockIdx.x * 4096 + (size_t)t * 16;
  float v[16];
  const float4* lp = (const float4*)(logt + base);
  float run = 0.f;
#pragma unroll
  for (int q = 0; q < 4; ++q) {
    float4 x = lp[q];
    run += x.x; v[q * 4 + 0] = run;
    run += x.y; v[q * 4 + 1] = run;
    run += x.z; v[q * 4 + 2] = run;
    run += x.w; v[q * 4 + 3] = run;
  }
  s[t] = run;
  __syncthreads();
  for (int off = 1; off < 256; off <<= 1) {
    float x = (t >= off) ? s[t - off] : 0.f;
    __syncthreads();
    s[t] += x;
    __syncthreads();
  }
  float excl = (t > 0) ? s[t - 1] : 0.f;
#pragma unroll
  for (int j = 0; j < 16; ++j) c[base + j] = v[j] + excl;
  if (t == 255) bsums[blockIdx.x] = s[255];
}

__global__ void scan2_kernel(const float* __restrict__ bsums, float* __restrict__ boff) {
  __shared__ float s[256];
  int t = threadIdx.x;
  s[t] = bsums[t];
  __syncthreads();
  for (int off = 1; off < 256; off <<= 1) {
    float x = (t >= off) ? s[t - off] : 0.f;
    __syncthreads();
    s[t] += x;
    __syncthreads();
  }
  boff[t] = (t > 0) ? s[t - 1] : 0.f;
}

__global__ void scan3_kernel(float* __restrict__ c, const float* __restrict__ boff) {
  int i = blockIdx.x * 256 + threadIdx.x;
  if (i < MTOT) c[i] += boff[i >> 12];
}

__global__ void raystart_kernel(const int* __restrict__ ray_id, const float* __restrict__ c,
                                const float* __restrict__ logt, float* __restrict__ start_off) {
  int i = blockIdx.x * 256 + threadIdx.x;
  if (i >= MTOT) return;
  int r = ray_id[i];
  if (i == 0 || ray_id[i - 1] != r) start_off[r] = c[i] - logt[i];
}

__global__ void weights_kernel(const int* __restrict__ ray_id, const float* __restrict__ c,
                               const float* __restrict__ logt, const float* __restrict__ alpha,
                               const float* __restrict__ start_off, float* __restrict__ wts,
                               float* __restrict__ out) {
  int i = blockIdx.x * 256 + threadIdx.x;
  if (i >= MTOT) return;
  int r = ray_id[i];
  float s0 = start_off[r];
  float e = c[i] - logt[i];
  wts[i] = alpha[i] * expf(e - s0);
  if (i == MTOT - 1 || ray_id[i + 1] != r) {   // last sample of ray -> alphainv_last
    float v = expf(c[i] - s0);
    out[r * 3 + 0] = v; out[r * 3 + 1] = v; out[r * 3 + 2] = v;
  }
}

// ---- fused 3-layer MLP on WMMA: 128-row tile per block, 16 rows per wave ----
// B fragments staged 4-at-a-time so one DS wait covers 4 back-to-back WMMAs.
__global__ void __launch_bounds__(256) mlp_kernel(
    const _Float16* __restrict__ feat, const float* __restrict__ wts,
    const int* __restrict__ ray_id,
    const _Float16* __restrict__ w0p, const _Float16* __restrict__ w1p,
    const _Float16* __restrict__ w2p,
    const float* __restrict__ b0, const float* __restrict__ b1, const float* __restrict__ b2,
    float* __restrict__ out) {
  __shared__ _Float16 sh_h[128 * 128];   // activation tile (f16)
  __shared__ _Float16 sh_w[128 * 128];   // weight staging (f16, fragment layout)

  const int t = threadIdx.x, lane = t & 31, wave = t >> 5;
  const int waveRow = wave * 16;
  const size_t mbase = (size_t)blockIdx.x * 128;
  const int rlo = waveRow + ((lane >> 4) & 1) * 8;   // D-layout row base
  const int col0 = lane & 15;

  // stage w0 (64x128 halves = 16KB)
  {
    const uint4* src = (const uint4*)w0p; uint4* dst = (uint4*)sh_w;
    for (int k = t; k < 1024; k += 256) dst[k] = src[k];
  }
  __syncthreads();

  v8f acc[8];
  const v8f vzero = {0.f, 0.f, 0.f, 0.f, 0.f, 0.f, 0.f, 0.f};
  v16h bf[4];

  // ---- layer 0: feat[128x64] @ w0[64x128] ----
#pragma unroll
  for (int nt = 0; nt < 8; ++nt) acc[nt] = vzero;
#pragma unroll
  for (int kt = 0; kt < 2; ++kt) {
    v16h a = load_a_frag(feat, 64, (int)mbase + waveRow, kt, lane);
#pragma unroll
    for (int h = 0; h < 2; ++h) {
#pragma unroll
      for (int j = 0; j < 4; ++j) bf[j] = load_b_frag(sh_w, 128, kt, h * 4 + j, lane);
#pragma unroll
      for (int j = 0; j < 4; ++j)
        acc[h * 4 + j] = __builtin_amdgcn_wmma_f32_16x16x32_f16(
            false, a, false, bf[j], (short)0, acc[h * 4 + j], false, false);
    }
  }
  __syncthreads();
  // stage w1 (32KB) while writing h = relu(acc + b0)
  {
    const uint4* src = (const uint4*)w1p; uint4* dst = (uint4*)sh_w;
    for (int k = t; k < 2048; k += 256) dst[k] = src[k];
  }
#pragma unroll
  for (int nt = 0; nt < 8; ++nt) {
    int col = nt * 16 + col0;
    float bias = b0[col];
#pragma unroll
    for (int v = 0; v < 8; ++v) {
      float x = acc[nt][v] + bias;
      sh_h[(rlo + v) * 128 + col] = (_Float16)(x > 0.f ? x : 0.f);
    }
  }
  __syncthreads();

  // ---- layer 1: h[128x128] @ w1[128x128] ----
#pragma unroll
  for (int nt = 0; nt < 8; ++nt) acc[nt] = vzero;
#pragma unroll
  for (int kt = 0; kt < 4; ++kt) {
    v16h a = load_a_frag(sh_h, 128, waveRow, kt, lane);
#pragma unroll
    for (int h = 0; h < 2; ++h) {
#pragma unroll
      for (int j = 0; j < 4; ++j) bf[j] = load_b_frag(sh_w, 128, kt, h * 4 + j, lane);
#pragma unroll
      for (int j = 0; j < 4; ++j)
        acc[h * 4 + j] = __builtin_amdgcn_wmma_f32_16x16x32_f16(
            false, a, false, bf[j], (short)0, acc[h * 4 + j], false, false);
    }
  }
  __syncthreads();
  // stage w2 (4KB) while writing h2 = relu(acc + b1)
  {
    const uint4* src = (const uint4*)w2p; uint4* dst = (uint4*)sh_w;
    for (int k = t; k < 256; k += 256) dst[k] = src[k];
  }
#pragma unroll
  for (int nt = 0; nt < 8; ++nt) {
    int col = nt * 16 + col0;
    float bias = b1[col];
#pragma unroll
    for (int v = 0; v < 8; ++v) {
      float x = acc[nt][v] + bias;
      sh_h[(rlo + v) * 128 + col] = (_Float16)(x > 0.f ? x : 0.f);
    }
  }
  __syncthreads();

  // ---- layer 2: h2[128x128] @ w2[128x16] (3 valid cols) ----
  // preload all 4 A and 4 B fragments, then 4 consecutive WMMAs
  v16h af2[4];
#pragma unroll
  for (int kt = 0; kt < 4; ++kt) {
    af2[kt] = load_a_frag(sh_h, 128, waveRow, kt, lane);
    bf[kt]  = load_b_frag(sh_w, 16, kt, 0, lane);
  }
  v8f r2 = vzero;
#pragma unroll
  for (int kt = 0; kt < 4; ++kt)
    r2 = __builtin_amdgcn_wmma_f32_16x16x32_f16(false, af2[kt], false, bf[kt], (short)0,
                                                r2, false, false);

  // ---- sigmoid + weighted segment-sum into out ----
  if (col0 < 3) {
    float bias = b2[col0];
#pragma unroll
    for (int v = 0; v < 8; ++v) {
      size_t gi = mbase + (size_t)(rlo + v);
      float rgb = 1.0f / (1.0f + expf(-(r2[v] + bias)));
      int r = ray_id[gi];
      atomicAdd(&out[r * 3 + col0], wts[gi] * rgb);
    }
  }
}

// ---------------- host launch ----------------
extern "C" void kernel_launch(void* const* d_in, const int* in_sizes, int n_in,
                              void* d_out, int out_size, void* d_ws, size_t ws_size,
                              hipStream_t stream) {
  const float* xyz       = (const float*)d_in[0];
  const float* viewdirs  = (const float*)d_in[1];
  const float* dgrid     = (const float*)d_in[2];
  const float* k0grid    = (const float*)d_in[3];
  const float* w0        = (const float*)d_in[4];
  const float* b0        = (const float*)d_in[5];
  const float* w1        = (const float*)d_in[6];
  const float* b1        = (const float*)d_in[7];
  const float* w2        = (const float*)d_in[8];
  const float* b2        = (const float*)d_in[9];
  const int*   ray_id    = (const int*)d_in[10];
  float* out = (float*)d_out;

  char* ws = (char*)d_ws;
  size_t off = 0;
  _Float16* feat = (_Float16*)(ws); off += (size_t)MTOT * 64 * sizeof(_Float16);
  float* alpha   = (float*)(ws + off);    off += (size_t)MTOT * sizeof(float);
  float* logt    = (float*)(ws + off);    off += (size_t)MTOT * sizeof(float);
  float* csum    = (float*)(ws + off);    off += (size_t)MTOT * sizeof(float);
  float* wts     = (float*)(ws + off);    off += (size_t)MTOT * sizeof(float);
  float* bsums   = (float*)(ws + off);    off += 256 * sizeof(float);
  float* boff    = (float*)(ws + off);    off += 256 * sizeof(float);
  float* soff    = (float*)(ws + off);    off += NRAYS * sizeof(float);
  float* vemb    = (float*)(ws + off);    off += (size_t)NRAYS * 27 * sizeof(float);
  _Float16* w0p  = (_Float16*)(ws + off); off += 64 * 128 * sizeof(_Float16);
  _Float16* w1p  = (_Float16*)(ws + off); off += 128 * 128 * sizeof(_Float16);
  _Float16* w2p  = (_Float16*)(ws + off); off += 128 * 16 * sizeof(_Float16);

  init_out_kernel    <<<48,   256, 0, stream>>>(out);
  vemb_kernel        <<<16,   256, 0, stream>>>(viewdirs, vemb);
  prep_weights_kernel<<<104,  256, 0, stream>>>(w0, w1, w2, w0p, w1p, w2p);
  sample_kernel      <<<MTOT / 256, 256, 0, stream>>>(xyz, dgrid, k0grid, ray_id, vemb,
                                                      feat, alpha, logt);
  scan1_kernel       <<<256,  256, 0, stream>>>(logt, csum, bsums);
  scan2_kernel       <<<1,    256, 0, stream>>>(bsums, boff);
  scan3_kernel       <<<MTOT / 256, 256, 0, stream>>>(csum, boff);
  raystart_kernel    <<<MTOT / 256, 256, 0, stream>>>(ray_id, csum, logt, soff);
  weights_kernel     <<<MTOT / 256, 256, 0, stream>>>(ray_id, csum, logt, alpha, soff, wts, out);
  mlp_kernel         <<<MTOT / 128, 256, 0, stream>>>(feat, wts, ray_id, w0p, w1p, w2p,
                                                      b0, b1, b2, out);
  (void)in_sizes; (void)n_in; (void)out_size; (void)ws_size;
}
